// NetAtom_13649406067125
// MI455X (gfx1250) — matrix operand order repaired and checked by reading, not compiled, not run on hardware
//
#include <hip/hip_runtime.h>

typedef __attribute__((ext_vector_type(16))) _Float16 v16h;
typedef __attribute__((ext_vector_type(4)))  _Float16 v4h;
typedef __attribute__((ext_vector_type(8)))  float    v8f;
typedef __attribute__((ext_vector_type(4)))  float    v4f;

#define D_IN 128   // descriptor size (K of layer 1)
#define H_W  256   // hidden width

// ---- branch-free transcendentals on the CDNA5 transcendental VALU ----
__device__ __forceinline__ float fast_tanh(float x) {
#if __has_builtin(__builtin_amdgcn_tanhf)
    return __builtin_amdgcn_tanhf(x);          // v_tanh_f32
#else
    // tanh(x) = sign(x) * (1 - 2/(exp2(2|x|*log2e)+1)); branch-free
    float ax = fabsf(x);
    float t  = __builtin_amdgcn_exp2f(ax * 2.8853900817779268f);   // v_exp_f32
    float r  = 1.0f - 2.0f * __builtin_amdgcn_rcpf(t + 1.0f);      // v_rcp_f32
    return copysignf(r, x);
#endif
}

__device__ __forceinline__ float fast_softplus(float x) {
    // softplus(x) = max(x,0) + log(1+exp(-|x|)); branch-free via exp2/log2
    float ax = fabsf(x);
    float e  = __builtin_amdgcn_exp2f(ax * -1.4426950408889634f);  // v_exp_f32
    float l  = __builtin_amdgcn_logf(1.0f + e) * 0.6931471805599453f; // v_log_f32
    return fmaxf(x, 0.0f) + l;
}

// ---------------------------------------------------------------------------
// MLP kernel: one block = 256 atoms (8 waves x 2 stripes x 16 atoms),
// blockIdx.y = species.
// LDS (dynamic, 260 KB):
//   [0      , 64KB )  W1^T f16 B-fragments: 64 frags (16 ntile x 4 kf) x 512 halfs
//   [64KB   , 192KB)  W2^T f16 B-fragments: 128 frags (16 ntile x 8 kf) x 512 halfs
//   [192KB  , 256KB)  h1 staging: 8 waves x (8 kf-frags x 512 halfs), A-frag layout
//   [256KB  , 260KB)  b1 (256 f32), b2 (256 f32), W3 (512 f32)
// Fragment layouts (wave32, v_wmma_f32_16x16x32_f16):
//   A: lane l -> M = l&15 ; half e -> K = 32*kf + ((e>>3)<<4) + ((l>>4)<<3) + (e&7)
//   B: lane l -> N = l&15 ; half e -> K = 32*kf + ((e>>3)<<4) + ((l>>4)<<3) + (e&7)   (same as A)
//   C/D: vgpr r, lane l -> (M = r + 8*(l>>4), N = l&15)
// ---------------------------------------------------------------------------
__global__ __launch_bounds__(256)
void bp_mlp_kernel(const float* __restrict__ desc0, const float* __restrict__ desc1,
                   const float* __restrict__ W1_0, const float* __restrict__ b1_0,
                   const float* __restrict__ W2_0, const float* __restrict__ b2_0,
                   const float* __restrict__ W3_0, const float* __restrict__ b3_0,
                   const float* __restrict__ W1_1, const float* __restrict__ b1_1,
                   const float* __restrict__ W2_1, const float* __restrict__ b2_1,
                   const float* __restrict__ W3_1, const float* __restrict__ b3_1,
                   float* __restrict__ m0, float* __restrict__ v0,
                   float* __restrict__ m1, float* __restrict__ v1,
                   int N)
{
    extern __shared__ char smem[];
    _Float16* w1f = (_Float16*)(smem);                     // 32768 halfs
    _Float16* w2f = (_Float16*)(smem + 65536);             // 65536 halfs
    _Float16* h1f = (_Float16*)(smem + 65536 + 131072);    // 32768 halfs
    float*    b1s = (float*)(smem + 262144);               // 256 f32
    float*    b2s = b1s + 256;                             // 256 f32
    float*    w3s = b2s + 256;                             // 512 f32

    const int sp = blockIdx.y;
    const float* desc = sp ? desc1 : desc0;
    const float* W1 = sp ? W1_1 : W1_0;  const float* b1v = sp ? b1_1 : b1_0;
    const float* W2 = sp ? W2_1 : W2_0;  const float* b2v = sp ? b2_1 : b2_0;
    const float* W3 = sp ? W3_1 : W3_0;  const float* b3v = sp ? b3_1 : b3_0;
    float* outM = sp ? m1 : m0;
    float* outV = sp ? v1 : v0;

    const int tid = threadIdx.x;

    // ---- stage W1^T into B-fragment order: coalesced float4 reads,
    //      packed ds_store_b64 writes (4 consecutive K share one lane-run) ----
    for (int i4 = tid; i4 < (H_W * D_IN) / 4; i4 += 256) {
        int base = i4 << 2;              // n*128 + K
        int n = base >> 7;
        int K = base & 127;
        float4 f = *(const float4*)(W1 + base);
        int frag = ((n >> 4) << 2) | (K >> 5);
        int lan  = (((K >> 3) & 1) << 4) | (n & 15);
        v4h h;
        h[0] = (_Float16)f.x; h[1] = (_Float16)f.y;
        h[2] = (_Float16)f.z; h[3] = (_Float16)f.w;
        *(v4h*)(w1f + frag * 512 + lan * 16 + ((((K >> 4) & 1) << 3) | (K & 7))) = h;
    }
    // ---- stage W2^T ----
    for (int i4 = tid; i4 < (H_W * H_W) / 4; i4 += 256) {
        int base = i4 << 2;              // n*256 + K
        int n = base >> 8;
        int K = base & 255;
        float4 f = *(const float4*)(W2 + base);
        int frag = ((n >> 4) << 3) | (K >> 5);
        int lan  = (((K >> 3) & 1) << 4) | (n & 15);
        v4h h;
        h[0] = (_Float16)f.x; h[1] = (_Float16)f.y;
        h[2] = (_Float16)f.z; h[3] = (_Float16)f.w;
        *(v4h*)(w2f + frag * 512 + lan * 16 + ((((K >> 4) & 1) << 3) | (K & 7))) = h;
    }
    // ---- stage small params ----
    b1s[tid] = b1v[tid];
    b2s[tid] = b2v[tid];
    w3s[tid]       = W3[tid];
    w3s[256 + tid] = W3[256 + tid];
    __syncthreads();

    const int lane = tid & 31;
    const int wv   = tid >> 5;
    const int g    = lane >> 4;     // half-wave group
    const int ln   = lane & 15;

    _Float16* h1w = h1f + wv * 4096;   // this wave's h1 staging (8 frags x 512)
    const float bm = b3v[0], bv = b3v[1];

    for (int s = 0; s < 2; ++s) {
        const int atomBase = blockIdx.x * 256 + wv * 32 + s * 16;

        // ---- layer-1 A fragments from global desc (f32 -> f16) ----
        int arow = atomBase + ln;
        arow = arow < N ? arow : (N - 1);
        const float* drow = desc + (size_t)arow * D_IN;

        v16h a1[4];
#pragma unroll
        for (int kf = 0; kf < 4; ++kf) {
            int base = kf * 32 + g * 8;
            float4 fa = *(const float4*)(drow + base);
            float4 fb = *(const float4*)(drow + base + 4);
            float4 fc = *(const float4*)(drow + base + 16);
            float4 fd = *(const float4*)(drow + base + 20);
            v16h a;
            a[0]  = (_Float16)fa.x; a[1]  = (_Float16)fa.y; a[2]  = (_Float16)fa.z; a[3]  = (_Float16)fa.w;
            a[4]  = (_Float16)fb.x; a[5]  = (_Float16)fb.y; a[6]  = (_Float16)fb.z; a[7]  = (_Float16)fb.w;
            a[8]  = (_Float16)fc.x; a[9]  = (_Float16)fc.y; a[10] = (_Float16)fc.z; a[11] = (_Float16)fc.w;
            a[12] = (_Float16)fd.x; a[13] = (_Float16)fd.y; a[14] = (_Float16)fd.z; a[15] = (_Float16)fd.w;
            a1[kf] = a;
        }

        // ---- layer 1: h1 = tanh(desc @ W1^T + b1) -> LDS (A-frag layout) ----
        for (int ntile = 0; ntile < 16; ++ntile) {
            float bias = b1s[ntile * 16 + ln];
            v8f acc;
#pragma unroll
            for (int r = 0; r < 8; ++r) acc[r] = bias;
            v16h bcur = *(const v16h*)(w1f + (ntile * 4) * 512 + lane * 16);
#pragma unroll
            for (int kf = 0; kf < 4; ++kf) {
                v16h bnext;
                if (kf < 3)
                    bnext = *(const v16h*)(w1f + (ntile * 4 + kf + 1) * 512 + lane * 16);
                acc = __builtin_amdgcn_wmma_f32_16x16x32_f16(
                    false, a1[kf], false, bcur, (short)0, acc, false, false);
                bcur = bnext;
            }
#pragma unroll
            for (int r = 0; r < 8; ++r) {
                float t = fast_tanh(acc[r]);
                int M   = r + g * 8;
                int K   = ntile * 16 + ln;       // layer-2 K coordinate
                int kf2 = K >> 5;
                int kk  = K & 31;
                int dl  = M | (((kk >> 3) & 1) << 4);
                int e   = ((kk >> 4) << 3) | (kk & 7);
                h1w[kf2 * 512 + dl * 16 + e] = (_Float16)t;
            }
        }

        // ---- layer-2 A fragments (same-wave LDS region; DS ops in-order) ----
        v16h a2[8];
#pragma unroll
        for (int kf = 0; kf < 8; ++kf)
            a2[kf] = *(const v16h*)(h1w + kf * 512 + lane * 16);

        // ---- layer 2 + fused layer 3 ----
        float accM[8], accV[8];
#pragma unroll
        for (int r = 0; r < 8; ++r) { accM[r] = 0.0f; accV[r] = 0.0f; }

        for (int ntile = 0; ntile < 16; ++ntile) {
            float bias = b2s[ntile * 16 + ln];
            v8f acc;
#pragma unroll
            for (int r = 0; r < 8; ++r) acc[r] = bias;
            v16h bcur = *(const v16h*)(w2f + (ntile * 8) * 512 + lane * 16);
#pragma unroll
            for (int kf = 0; kf < 8; ++kf) {
                v16h bnext;
                if (kf < 7)
                    bnext = *(const v16h*)(w2f + (ntile * 8 + kf + 1) * 512 + lane * 16);
                acc = __builtin_amdgcn_wmma_f32_16x16x32_f16(
                    false, a2[kf], false, bcur, (short)0, acc, false, false);
                bcur = bnext;
            }
            int col = ntile * 16 + ln;
            float w30 = w3s[col];
            float w31 = w3s[H_W + col];
#pragma unroll
            for (int r = 0; r < 8; ++r) {
                float t  = fast_tanh(acc[r]);
                float c0 = t * w30;
                float c1 = t * w31;
                // reduce across each 16-lane half (N dimension)
#pragma unroll
                for (int mask = 8; mask >= 1; mask >>= 1) {
                    c0 += __shfl_xor(c0, mask, 32);
                    c1 += __shfl_xor(c1, mask, 32);
                }
                accM[r] += c0;
                accV[r] += c1;
            }
        }

        // lanes 0 and 16 hold the per-atom sums (M = r + 8*g)
        if (ln == 0) {
#pragma unroll
            for (int r = 0; r < 8; ++r) {
                int atom = atomBase + g * 8 + r;
                if (atom < N) {
                    outM[atom] = accM[r] + bm;
                    outV[atom] = fast_softplus(accV[r] + bv);
                }
            }
        }
    }
}

// ---------------------------------------------------------------------------
// Reduction: out[k] = logic0[k,:]@m0 + logic1[k,:]@m1 ; out[K+k] = same with v.
// One block per structure k; logic rows streamed once with non-temporal b128
// loads (protects L2-resident m/v vectors), plus global_prefetch ahead.
// ---------------------------------------------------------------------------
__global__ __launch_bounds__(256)
void bp_reduce_kernel(const float* __restrict__ logic0, const float* __restrict__ logic1,
                      const float* __restrict__ m0, const float* __restrict__ v0,
                      const float* __restrict__ m1, const float* __restrict__ v1,
                      float* __restrict__ out, int N, int K)
{
    const int k = blockIdx.x;
    const int t = threadIdx.x;
    const float* r0 = logic0 + (size_t)k * N;
    const float* r1 = logic1 + (size_t)k * N;

    float am = 0.0f, av = 0.0f;
    const int nv = N >> 2;
    const v4f* r04 = (const v4f*)r0;
    const v4f* r14 = (const v4f*)r1;
    const v4f* m04 = (const v4f*)m0;
    const v4f* v04 = (const v4f*)v0;
    const v4f* m14 = (const v4f*)m1;
    const v4f* v14 = (const v4f*)v1;

    for (int i = t; i < nv; i += 256) {
        __builtin_prefetch(r04 + i + 2048, 0, 0);   // global_prefetch ~32KB ahead
        __builtin_prefetch(r14 + i + 2048, 0, 0);
        v4f a = __builtin_nontemporal_load(r04 + i);
        v4f b = __builtin_nontemporal_load(r14 + i);
        v4f x = m04[i], y = v04[i];
        v4f p = m14[i], q = v14[i];
#pragma unroll
        for (int c = 0; c < 4; ++c) {
            am += a[c] * x[c] + b[c] * p[c];
            av += a[c] * y[c] + b[c] * q[c];
        }
    }
    for (int i = (nv << 2) + t; i < N; i += 256) {   // tail (N % 4)
        am += r0[i] * m0[i] + r1[i] * m1[i];
        av += r0[i] * v0[i] + r1[i] * v1[i];
    }

    // wave32 reduce
#pragma unroll
    for (int off = 16; off >= 1; off >>= 1) {
        am += __shfl_down(am, off, 32);
        av += __shfl_down(av, off, 32);
    }
    __shared__ float sm[8], sv[8];
    if ((t & 31) == 0) { sm[t >> 5] = am; sv[t >> 5] = av; }
    __syncthreads();
    if (t == 0) {
        float tm = 0.0f, tv = 0.0f;
#pragma unroll
        for (int i = 0; i < 8; ++i) { tm += sm[i]; tv += sv[i]; }
        out[k]     = tm;
        out[K + k] = tv;
    }
}

extern "C" void kernel_launch(void* const* d_in, const int* in_sizes, int n_in,
                              void* d_out, int out_size, void* d_ws, size_t ws_size,
                              hipStream_t stream) {
    const float* desc0  = (const float*)d_in[0];
    const float* desc1  = (const float*)d_in[1];
    const float* logic0 = (const float*)d_in[2];
    const float* logic1 = (const float*)d_in[3];
    const float* W1_0 = (const float*)d_in[4];  const float* b1_0 = (const float*)d_in[5];
    const float* W2_0 = (const float*)d_in[6];  const float* b2_0 = (const float*)d_in[7];
    const float* W3_0 = (const float*)d_in[8];  const float* b3_0 = (const float*)d_in[9];
    const float* W1_1 = (const float*)d_in[10]; const float* b1_1 = (const float*)d_in[11];
    const float* W2_1 = (const float*)d_in[12]; const float* b2_1 = (const float*)d_in[13];
    const float* W3_1 = (const float*)d_in[14]; const float* b3_1 = (const float*)d_in[15];

    const int N = in_sizes[0] / D_IN;       // 100000
    const int K = in_sizes[2] / N;          // 1000

    float* m0 = (float*)d_ws;
    float* v0 = m0 + N;
    float* m1 = v0 + N;
    float* v1 = m1 + N;

    dim3 grid1((N + 255) / 256, 2);
    size_t shmem = 65536 + 131072 + 65536 + 4096;  // 260 KB dynamic LDS
    bp_mlp_kernel<<<grid1, 256, shmem, stream>>>(
        desc0, desc1,
        W1_0, b1_0, W2_0, b2_0, W3_0, b3_0,
        W1_1, b1_1, W2_1, b2_1, W3_1, b3_1,
        m0, v0, m1, v1, N);

    bp_reduce_kernel<<<K, 256, 0, stream>>>(logic0, logic1, m0, v0, m1, v1,
                                            (float*)d_out, N, K);
}
